// LaSPNet_85023172591738
// MI455X (gfx1250) — matrix operand order repaired and checked by reading, not compiled, not run on hardware
//
// MI455X / gfx1250 implementation of LaSPNet (CNN head + PCG solve).
//
// Precision/data-movement rationale (no runtime available — reasoned from specs):
//  * Convs = ~100 GFLOP of GEMM-like work -> v_wmma_f32_16x16x32_f16 (matrix path),
//    activations stored NHWC f16 so A-fragments are two b128 loads per lane.
//    Weights pre-packed to a contiguous-k B-fragment lane layout and staged via LDS.
//  * PCG = ~6 GB of f32 streaming over 200 iterations -> bandwidth bound
//    (~0.26 ms at 23.3 TB/s). Fused stencil kernels, deterministic two-stage
//    reductions (graph replay must be bitwise stable -> no float atomics).
//  * Workspace use ~172 MB (acts ping-pong 2x64MB f16, 14 f32 fields, packed W).

#include <hip/hip_runtime.h>
#include <hip/hip_bf16.h>

typedef __attribute__((ext_vector_type(16))) _Float16 v16h;
typedef __attribute__((ext_vector_type(8)))  _Float16 v8h;
typedef __attribute__((ext_vector_type(8)))  float    v8f;

#define BATCH 2
#define HH 512
#define WW 512
#define HWP (HH*WW)
#define NPIX (BATCH*HWP)
#define NBLK (NPIX/256)   // 2048 blocks of 256 threads cover the grid exactly

// ---------------------------------------------------------------- helpers
__device__ __forceinline__ float softplus_f(float x) {
  // log1p(exp(x)) stable
  return fmaxf(x, 0.0f) + log1pf(expf(-fabsf(x)));
}

__device__ __forceinline__ float block_reduce256(float v) {
  __shared__ float sh[256];
  int t = threadIdx.x;
  sh[t] = v;
  __syncthreads();
#pragma unroll
  for (int s = 128; s > 0; s >>= 1) {
    if (t < s) sh[t] += sh[t + s];
    __syncthreads();
  }
  float r = sh[0];
  __syncthreads();   // protect sh before any subsequent call
  return r;
}

// 5-point operator: L(u) + (lam+muML)*u. Boundary weights are exactly zero
// (built that way), so skipping out-of-range neighbors matches jnp.roll+zeroed-w.
__device__ __forceinline__ float apply_op(
    const float* __restrict__ u,
    const float* __restrict__ wE, const float* __restrict__ wWv,
    const float* __restrict__ wS, const float* __restrict__ wN,
    const float* __restrict__ sA, int idx, int x, int y) {
  float uc  = u[idx];
  float acc = sA[idx] * uc;
  if (x < WW - 1) acc += wE[idx]  * (u[idx + 1]  - uc);
  if (x > 0)      acc += wWv[idx] * (u[idx - 1]  - uc);
  if (y < HH - 1) acc += wS[idx]  * (u[idx + WW] - uc);
  if (y > 0)      acc += wN[idx]  * (u[idx - WW] - uc);
  return acc;
}

// ---------------------------------------------------------------- weight packing
// B-fragment layout used here (contiguous k, same as A): per fragment 32 lanes x 16 f16,
// lane<16 -> N=lane, K=j ; lane>=16 -> N=lane-16, K=16+j. Stored lane-major.

__global__ void pack_conv_kernel(const float* __restrict__ w, _Float16* __restrict__ out,
                                 int Cin, int Cout) {
  int KC = Cin / 32, NB = Cout / 16;
  int total = 9 * KC * NB * 512;
  for (int i = blockIdx.x * blockDim.x + threadIdx.x; i < total;
       i += gridDim.x * blockDim.x) {
    int j  = i & 15;
    int l  = (i >> 4) & 31;
    int fi = i >> 9;                 // fragment = (t*KC+kc)*NB+nb
    int nb = fi % NB;
    int rest = fi / NB;
    int kc = rest % KC;
    int t  = rest / KC;              // tap 0..8 (ky*3+kx)
    int n  = nb * 16 + (l & 15);
    int cin = kc * 32 + (l >> 4) * 16 + j;
    out[i] = (_Float16)w[((size_t)n * Cin + cin) * 9 + t];
  }
}

// conv1: K chunks pack 4 taps x 8 padded channels (7 real + 1 zero).
__global__ void pack_conv1_kernel(const float* __restrict__ w1, _Float16* __restrict__ out) {
  int total = 12 * 512;              // 3 chunks x 4 nb fragments
  for (int i = blockIdx.x * blockDim.x + threadIdx.x; i < total;
       i += gridDim.x * blockDim.x) {
    int j  = i & 15;
    int l  = (i >> 4) & 31;
    int fi = i >> 9;
    int nb = fi & 3;
    int cc = fi >> 2;
    int K  = (l >> 4) * 16 + j;      // 0..31 within chunk
    int tap = cc * 4 + (K >> 3);
    int cin = K & 7;
    int n   = nb * 16 + (l & 15);
    float v = 0.0f;
    if (tap < 9 && cin < 7) v = w1[((size_t)n * 7 + cin) * 9 + tap];
    out[i] = (_Float16)v;
  }
}

// ---------------------------------------------------------------- input assembly
// x7: NHWC f16, C=8 : [I0,I1,I2,Pn,E,MLf,edgeP,0]
__global__ void assemble_x7_kernel(const float* __restrict__ I, const float* __restrict__ P,
                                   const float* __restrict__ E, const int* __restrict__ ML,
                                   _Float16* __restrict__ x7) {
  int idx = blockIdx.x * 256 + threadIdx.x;
  int x = idx & (WW - 1);
  int y = (idx >> 9) & (HH - 1);
  int b = idx >> 18;
  float p  = P[idx];
  float pn = fminf(fmaxf(p * 0.1f, 0.0f), 1.0f);
  float gx = 0.0f, gy = 0.0f;
  if (x > 0) gx = pn - fminf(fmaxf(P[idx - 1]  * 0.1f, 0.0f), 1.0f);
  if (y > 0) gy = pn - fminf(fmaxf(P[idx - WW] * 0.1f, 0.0f), 1.0f);
  float edge = 0.5f * (fabsf(gx) + fabsf(gy));
  float mlf  = (ML[idx] > 0) ? 1.0f : 0.0f;
  size_t base = (size_t)idx * 8;
  size_t cb   = (size_t)b * 3 * HWP + (size_t)y * WW + x;   // NCHW source
  x7[base + 0] = (_Float16)I[cb];
  x7[base + 1] = (_Float16)I[cb + HWP];
  x7[base + 2] = (_Float16)I[cb + 2 * HWP];
  x7[base + 3] = (_Float16)pn;
  x7[base + 4] = (_Float16)E[idx];
  x7[base + 5] = (_Float16)mlf;
  x7[base + 6] = (_Float16)edge;
  x7[base + 7] = (_Float16)0.0f;
}

// ---------------------------------------------------------------- conv layers
// Wave tile: M=16 pixels along W, N=NB*16 output channels; block = 8 waves = 128 px.

__global__ __launch_bounds__(256) void conv1_kernel(
    const _Float16* __restrict__ x7, const _Float16* __restrict__ wfrag,
    const float* __restrict__ bias, _Float16* __restrict__ out) {
  __shared__ __align__(32) _Float16 lw[12 * 512];   // 12 KB: all conv1 fragments
  int tid = threadIdx.x;
  {
    const uint4* g = (const uint4*)wfrag;
    uint4* d = (uint4*)lw;
    for (int i = tid; i < 12 * 512 / 8; i += 256) d[i] = g[i];
  }
  __syncthreads();
  int wave = tid >> 5, lane = tid & 31, hi = lane >> 4, m = lane & 15;
  int b = blockIdx.z, y = blockIdx.y;
  int px0 = blockIdx.x * 128 + wave * 16;
  int px  = px0 + m;
  v8f zf = {0, 0, 0, 0, 0, 0, 0, 0};
  v8f acc[4];
#pragma unroll
  for (int n = 0; n < 4; n++) acc[n] = zf;
#pragma unroll
  for (int cc = 0; cc < 3; cc++) {
    // lane half picks tap: low lanes K{0..15} -> taps cc*4+0 / cc*4+1
    int t0 = cc * 4 + 2 * hi;
    int t1 = cc * 4 + 2 * hi + 1;
    v8h va, vb;
#pragma unroll
    for (int i = 0; i < 8; i++) { va[i] = (_Float16)0.0f; vb[i] = (_Float16)0.0f; }
    if (t0 < 9) {
      int yy = y + t0 / 3 - 1, xx = px + t0 % 3 - 1;
      if (yy >= 0 && yy < HH && xx >= 0 && xx < WW)
        va = *(const v8h*)(x7 + (((size_t)b * HH + yy) * WW + xx) * 8);
    }
    if (t1 < 9) {
      int yy = y + t1 / 3 - 1, xx = px + t1 % 3 - 1;
      if (yy >= 0 && yy < HH && xx >= 0 && xx < WW)
        vb = *(const v8h*)(x7 + (((size_t)b * HH + yy) * WW + xx) * 8);
    }
    v16h a;
#pragma unroll
    for (int i = 0; i < 8; i++) { a[i] = va[i]; a[8 + i] = vb[i]; }
#pragma unroll
    for (int n = 0; n < 4; n++) {
      const v16h bf = *(const v16h*)(lw + ((size_t)(cc * 4 + n) * 32 + lane) * 16);
      acc[n] = __builtin_amdgcn_wmma_f32_16x16x32_f16(false, a, false, bf,
                                                      (short)0, acc[n], false, false);
    }
  }
#pragma unroll
  for (int n = 0; n < 4; n++) {
    float bv = bias[n * 16 + m];
#pragma unroll
    for (int j = 0; j < 8; j++) {
      int mm = hi * 8 + j;
      float v = fmaxf(acc[n][j] + bv, 0.0f);
      out[(((size_t)b * HH + y) * WW + (px0 + mm)) * 64 + n * 16 + m] = (_Float16)v;
    }
  }
}

template <int NB>
__global__ __launch_bounds__(256) void conv64_kernel(
    const _Float16* __restrict__ in, const _Float16* __restrict__ wfrag,
    const float* __restrict__ bias, _Float16* __restrict__ out) {
  constexpr int NFROW = 3 * 2 * NB;                     // frags per ky row
  __shared__ __align__(32) _Float16 lw[NFROW * 512];    // <=24 KB
  int tid = threadIdx.x, wave = tid >> 5, lane = tid & 31, hi = lane >> 4, m = lane & 15;
  int b = blockIdx.z, y = blockIdx.y;
  int px0 = blockIdx.x * 128 + wave * 16;
  int px  = px0 + m;
  v8f zf = {0, 0, 0, 0, 0, 0, 0, 0};
  v8f acc[NB];
#pragma unroll
  for (int n = 0; n < NB; n++) acc[n] = zf;

  for (int ky = 0; ky < 3; ky++) {
    __syncthreads();
    {
      const uint4* g = (const uint4*)(wfrag + (size_t)ky * NFROW * 512);
      uint4* d = (uint4*)lw;
      for (int i = tid; i < NFROW * 512 / 8; i += 256) d[i] = g[i];
    }
    __syncthreads();
    int yy = y + ky - 1;
    bool yok = (yy >= 0 && yy < HH);
#pragma unroll
    for (int tx = 0; tx < 3; tx++) {
      int xx = px + tx - 1;
      bool inb = yok && (xx >= 0) && (xx < WW);
#pragma unroll
      for (int kc = 0; kc < 2; kc++) {
        v16h a;
        if (inb) {
          const _Float16* src =
              in + (((size_t)b * HH + yy) * WW + xx) * 64 + kc * 32 + hi * 16;
          v8h lo  = *(const v8h*)src;
          v8h hi8 = *(const v8h*)(src + 8);
#pragma unroll
          for (int i = 0; i < 8; i++) { a[i] = lo[i]; a[8 + i] = hi8[i]; }
        } else {
#pragma unroll
          for (int i = 0; i < 16; i++) a[i] = (_Float16)0.0f;
        }
#pragma unroll
        for (int n = 0; n < NB; n++) {
          const v16h bf =
              *(const v16h*)(lw + ((size_t)((tx * 2 + kc) * NB + n) * 32 + lane) * 16);
          acc[n] = __builtin_amdgcn_wmma_f32_16x16x32_f16(false, a, false, bf,
                                                          (short)0, acc[n], false, false);
        }
      }
    }
  }
  const int Cout = NB * 16;
#pragma unroll
  for (int n = 0; n < NB; n++) {
    float bv = bias[n * 16 + m];
#pragma unroll
    for (int j = 0; j < 8; j++) {
      int mm = hi * 8 + j;
      float v = fmaxf(acc[n][j] + bv, 0.0f);
      out[(((size_t)b * HH + y) * WW + (px0 + mm)) * Cout + n * 16 + m] = (_Float16)v;
    }
  }
}

// ---------------------------------------------------------------- head + fields
__global__ void head_kernel(const _Float16* __restrict__ h, const float* __restrict__ w2h,
                            const float* __restrict__ b2h, const int* __restrict__ ML,
                            const float* __restrict__ P, const float* __restrict__ DL,
                            float* __restrict__ ax, float* __restrict__ ay,
                            float* __restrict__ sArr, float* __restrict__ rhs) {
  int idx = blockIdx.x * 256 + threadIdx.x;
  const _Float16* hp = h + (size_t)idx * 32;
  float r[4];
#pragma unroll
  for (int o = 0; o < 4; o++) {
    float accv = b2h[o];
    const float* wp = w2h + o * 32;
#pragma unroll
    for (int c = 0; c < 32; c++) accv += (float)hp[c] * wp[c];
    r[o] = accv;
  }
  float axv = softplus_f(r[0]) + 1e-4f;
  float ayv = softplus_f(r[1]) + 1e-4f;
  float lam = softplus_f(r[2]) + 1e-3f;
  float mu  = softplus_f(r[3]);
  float mlf  = (ML[idx] > 0) ? 1.0f : 0.0f;
  float muML = mu * mlf;
  ax[idx] = axv;
  ay[idx] = ayv;
  sArr[idx] = lam + muML;
  rhs[idx]  = lam * P[idx] + muML * DL[idx];
}

__global__ void build_weights_kernel(const float* __restrict__ ax,
                                     const float* __restrict__ ay,
                                     const float* __restrict__ sArr,
                                     float* __restrict__ wE, float* __restrict__ wWv,
                                     float* __restrict__ wS, float* __restrict__ wN,
                                     float* __restrict__ Minv) {
  int idx = blockIdx.x * 256 + threadIdx.x;
  int x = idx & (WW - 1);
  int y = (idx >> 9) & (HH - 1);
  float a0 = ax[idx], b0 = ay[idx];
  float e  = (x < WW - 1) ? 0.5f * (a0 + ax[idx + 1])  : 0.0f;
  float w_ = (x > 0)      ? 0.5f * (a0 + ax[idx - 1])  : 0.0f;
  float s_ = (y < HH - 1) ? 0.5f * (b0 + ay[idx + WW]) : 0.0f;
  float n_ = (y > 0)      ? 0.5f * (b0 + ay[idx - WW]) : 0.0f;
  wE[idx] = e; wWv[idx] = w_; wS[idx] = s_; wN[idx] = n_;
  Minv[idx] = 1.0f / (e + w_ + s_ + n_ + sArr[idx] + 1e-6f);
}

// ---------------------------------------------------------------- PCG
// Scalars S[]: 0=rz 1=pAp 2=alpha 3=beta 4=rr 5=normb2 6=done

__global__ void pcg_init_kernel(const float* __restrict__ P, const float* __restrict__ rhs,
                                const float* __restrict__ wE, const float* __restrict__ wWv,
                                const float* __restrict__ wS, const float* __restrict__ wN,
                                const float* __restrict__ sArr, const float* __restrict__ Minv,
                                float* __restrict__ xv, float* __restrict__ rv,
                                float* __restrict__ zv, float* __restrict__ pv,
                                float* __restrict__ part1, float* __restrict__ part2) {
  int idx = blockIdx.x * 256 + threadIdx.x;
  int x = idx & (WW - 1);
  int y = (idx >> 9) & (HH - 1);
  float x0 = P[idx];
  xv[idx] = x0;
  float A0 = apply_op(P, wE, wWv, wS, wN, sArr, idx, x, y);
  float bb = rhs[idx];
  float r  = bb - A0;
  rv[idx] = r;
  float z = Minv[idx] * r;
  zv[idx] = z;
  pv[idx] = z;
  float s1 = block_reduce256(r * z);
  float s2 = block_reduce256(bb * bb);
  if (threadIdx.x == 0) { part1[blockIdx.x] = s1; part2[blockIdx.x] = s2; }
}

__global__ void pcg_red_init_kernel(const float* __restrict__ p1,
                                    const float* __restrict__ p2, float* __restrict__ S) {
  float a = 0.0f, b = 0.0f;
  for (int k = threadIdx.x; k < NBLK; k += 256) { a += p1[k]; b += p2[k]; }
  a = block_reduce256(a);
  b = block_reduce256(b);
  if (threadIdx.x == 0) {
    S[0] = a; S[5] = b;
    S[1] = 0.0f; S[2] = 0.0f; S[3] = 0.0f; S[4] = 0.0f; S[6] = 0.0f;
  }
}

__global__ void pcg_apply_kernel(const float* __restrict__ pv, const float* __restrict__ wE,
                                 const float* __restrict__ wWv, const float* __restrict__ wS,
                                 const float* __restrict__ wN, const float* __restrict__ sArr,
                                 float* __restrict__ Apv, float* __restrict__ part1,
                                 const float* __restrict__ S) {
  if (S[6] != 0.0f) return;   // uniform across grid
  int idx = blockIdx.x * 256 + threadIdx.x;
  int x = idx & (WW - 1);
  int y = (idx >> 9) & (HH - 1);
  float v = apply_op(pv, wE, wWv, wS, wN, sArr, idx, x, y);
  Apv[idx] = v;
  float s1 = block_reduce256(pv[idx] * v);
  if (threadIdx.x == 0) part1[blockIdx.x] = s1;
}

__global__ void pcg_red_alpha_kernel(const float* __restrict__ p1, float* __restrict__ S) {
  if (S[6] != 0.0f) return;
  float a = 0.0f;
  for (int k = threadIdx.x; k < NBLK; k += 256) a += p1[k];
  a = block_reduce256(a);
  if (threadIdx.x == 0) { S[1] = a; S[2] = S[0] / (a + 1e-12f); }
}

__global__ void pcg_update_kernel(float* __restrict__ xv, float* __restrict__ rv,
                                  float* __restrict__ zv, const float* __restrict__ pv,
                                  const float* __restrict__ Apv, const float* __restrict__ Minv,
                                  float* __restrict__ part1, float* __restrict__ part2,
                                  const float* __restrict__ S) {
  if (S[6] != 0.0f) return;
  int idx = blockIdx.x * 256 + threadIdx.x;
  float alpha = S[2];
  float xn = xv[idx] + alpha * pv[idx];
  xv[idx] = xn;
  float rn = rv[idx] - alpha * Apv[idx];
  rv[idx] = rn;
  float zn = Minv[idx] * rn;
  zv[idx] = zn;
  float s1 = block_reduce256(rn * zn);
  float s2 = block_reduce256(rn * rn);
  if (threadIdx.x == 0) { part1[blockIdx.x] = s1; part2[blockIdx.x] = s2; }
}

__global__ void pcg_red_beta_kernel(const float* __restrict__ p1,
                                    const float* __restrict__ p2, float* __restrict__ S) {
  if (S[6] != 0.0f) return;
  float a = 0.0f, b = 0.0f;
  for (int k = threadIdx.x; k < NBLK; k += 256) { a += p1[k]; b += p2[k]; }
  a = block_reduce256(a);
  b = block_reduce256(b);
  if (threadIdx.x == 0) {
    S[3] = a / (S[0] + 1e-12f);    // beta
    S[0] = a;                      // rz <- rz_new
    S[4] = b;                      // ||r||^2
    float relres = sqrtf(b) / (sqrtf(S[5]) + 1e-12f);
    if (relres < 1e-5f) S[6] = 1.0f;
  }
}

__global__ void pcg_pupdate_kernel(float* __restrict__ pv, const float* __restrict__ zv,
                                   const float* __restrict__ S) {
  if (S[6] != 0.0f) return;
  int idx = blockIdx.x * 256 + threadIdx.x;
  pv[idx] = zv[idx] + S[3] * pv[idx];
}

__global__ void final_kernel(const float* __restrict__ xv, const float* __restrict__ P,
                             float* __restrict__ D) {
  int idx = blockIdx.x * 256 + threadIdx.x;
  float p = P[idx];
  float d = xv[idx] - p;
  d = fminf(fmaxf(d, -0.2f), 0.2f);
  D[idx] = fminf(fmaxf(p + d, 0.0f), 10.0f);
}

// ---------------------------------------------------------------- launcher
extern "C" void kernel_launch(void* const* d_in, const int* in_sizes, int n_in,
                              void* d_out, int out_size, void* d_ws, size_t ws_size,
                              hipStream_t stream) {
  (void)in_sizes; (void)n_in; (void)out_size; (void)ws_size;
  const float* I   = (const float*)d_in[0];
  const float* DL  = (const float*)d_in[1];
  const int*   ML  = (const int*)d_in[2];
  const float* E   = (const float*)d_in[3];
  const float* P   = (const float*)d_in[4];
  const float* w1  = (const float*)d_in[5];
  const float* b1  = (const float*)d_in[6];
  const float* w2  = (const float*)d_in[7];
  const float* b2  = (const float*)d_in[8];
  const float* w3  = (const float*)d_in[9];
  const float* b3  = (const float*)d_in[10];
  const float* hw1 = (const float*)d_in[11];
  const float* hb1 = (const float*)d_in[12];
  const float* hw2 = (const float*)d_in[13];
  const float* hb2 = (const float*)d_in[14];
  float* D = (float*)d_out;

  char* base = (char*)d_ws;
  size_t off = 0;
  auto carve = [&](size_t bytes) -> void* {
    void* p = base + off;
    off += (bytes + 255) & ~(size_t)255;
    return p;
  };
  _Float16* x7   = (_Float16*)carve((size_t)NPIX * 8 * 2);
  _Float16* bufA = (_Float16*)carve((size_t)NPIX * 64 * 2);
  _Float16* bufB = (_Float16*)carve((size_t)NPIX * 64 * 2);   // h32 reuses this
  _Float16* w1p  = (_Float16*)carve((size_t)12 * 512 * 2);
  _Float16* w2p  = (_Float16*)carve((size_t)72 * 512 * 2);
  _Float16* w3p  = (_Float16*)carve((size_t)72 * 512 * 2);
  _Float16* h1p  = (_Float16*)carve((size_t)36 * 512 * 2);
  float* ax   = (float*)carve((size_t)NPIX * 4);
  float* ay   = (float*)carve((size_t)NPIX * 4);
  float* sArr = (float*)carve((size_t)NPIX * 4);
  float* rhs  = (float*)carve((size_t)NPIX * 4);
  float* wE   = (float*)carve((size_t)NPIX * 4);
  float* wWv  = (float*)carve((size_t)NPIX * 4);
  float* wS   = (float*)carve((size_t)NPIX * 4);
  float* wN   = (float*)carve((size_t)NPIX * 4);
  float* Minv = (float*)carve((size_t)NPIX * 4);
  float* xv   = (float*)carve((size_t)NPIX * 4);
  float* rv   = (float*)carve((size_t)NPIX * 4);
  float* zv   = (float*)carve((size_t)NPIX * 4);
  float* pv   = (float*)carve((size_t)NPIX * 4);
  float* Apv  = (float*)carve((size_t)NPIX * 4);
  float* part1 = (float*)carve((size_t)NBLK * 4);
  float* part2 = (float*)carve((size_t)NBLK * 4);
  float* S     = (float*)carve(16 * 4);

  // 1) pack weights into WMMA B-fragment layout
  pack_conv1_kernel<<<24, 256, 0, stream>>>(w1, w1p);
  pack_conv_kernel<<<144, 256, 0, stream>>>(w2, w2p, 64, 64);
  pack_conv_kernel<<<144, 256, 0, stream>>>(w3, w3p, 64, 64);
  pack_conv_kernel<<<72, 256, 0, stream>>>(hw1, h1p, 64, 32);

  // 2) assemble 7(+1)-channel f16 input
  assemble_x7_kernel<<<NBLK, 256, 0, stream>>>(I, P, E, ML, x7);

  // 3) conv stack (WMMA)
  dim3 cgrid(WW / 128, HH, BATCH);
  conv1_kernel<<<cgrid, 256, 0, stream>>>(x7, w1p, b1, bufA);
  conv64_kernel<4><<<cgrid, 256, 0, stream>>>(bufA, w2p, b2, bufB);
  conv64_kernel<4><<<cgrid, 256, 0, stream>>>(bufB, w3p, b3, bufA);
  conv64_kernel<2><<<cgrid, 256, 0, stream>>>(bufA, h1p, hb1, bufB);  // -> C=32

  // 4) head 1x1 conv + softplus + field/rhs construction
  head_kernel<<<NBLK, 256, 0, stream>>>(bufB, hw2, hb2, ML, P, DL, ax, ay, sArr, rhs);
  build_weights_kernel<<<NBLK, 256, 0, stream>>>(ax, ay, sArr, wE, wWv, wS, wN, Minv);

  // 5) PCG (fixed 200-iteration schedule, device-side early-exit flag)
  pcg_init_kernel<<<NBLK, 256, 0, stream>>>(P, rhs, wE, wWv, wS, wN, sArr, Minv,
                                            xv, rv, zv, pv, part1, part2);
  pcg_red_init_kernel<<<1, 256, 0, stream>>>(part1, part2, S);
  for (int it = 0; it < 200; ++it) {
    pcg_apply_kernel<<<NBLK, 256, 0, stream>>>(pv, wE, wWv, wS, wN, sArr, Apv, part1, S);
    pcg_red_alpha_kernel<<<1, 256, 0, stream>>>(part1, S);
    pcg_update_kernel<<<NBLK, 256, 0, stream>>>(xv, rv, zv, pv, Apv, Minv, part1, part2, S);
    pcg_red_beta_kernel<<<1, 256, 0, stream>>>(part1, part2, S);
    pcg_pupdate_kernel<<<NBLK, 256, 0, stream>>>(pv, zv, S);
  }

  // 6) clamp and output
  final_kernel<<<NBLK, 256, 0, stream>>>(xv, P, D);
}
